// Init_Layer_44470091383040
// MI455X (gfx1250) — matrix operand order repaired and checked by reading, not compiled, not run on hardware
//
#include <hip/hip_runtime.h>
#include <hip/hip_bf16.h>

typedef __attribute__((ext_vector_type(16))) __bf16 v16bf;
typedef __attribute__((ext_vector_type(8)))  __bf16 v8bf;
typedef __attribute__((ext_vector_type(8)))  float  v8f;

#define DIM 256

// ---------------------------------------------------------------------------
// f32 -> bf16 conversion
// ---------------------------------------------------------------------------
__global__ __launch_bounds__(256) void k_cvt_bf16(const float* __restrict__ src,
                                                  __bf16* __restrict__ dst,
                                                  long long n) {
  long long i = (long long)blockIdx.x * blockDim.x + threadIdx.x;
  if (i < n) dst[i] = (__bf16)src[i];
}

// ---------------------------------------------------------------------------
// Y(MxDIM) = Xbf16(MxDIM) @ W(DIM x DIM, row-major, row = output feature)^T + b
// Block = 16 rows of X, 4 waves; wave w computes the 64-wide N range [64w, 64w+64)
// as four 16x16 WMMA tiles sharing one A fragment per K-step (2.5 b128 loads
// per v_wmma). Bias is pre-loaded into the f32 accumulators (C/D layout keeps a
// single output column per lane, so all 8 acc elements take the same bias).
// WMMA operand layouts follow CDNA5 ISA 7.12.2 (wave32, 16-bit A 16x32, B 32x16,
// f32 C/D 16x16).
// ---------------------------------------------------------------------------
__global__ __launch_bounds__(128)
void k_gemm_wmma_bf16(const __bf16* __restrict__ X, const __bf16* __restrict__ W,
                      const float* __restrict__ bias, float* __restrict__ Y, int M) {
  const int lane  = threadIdx.x & 31;
  const int wv    = threadIdx.x >> 5;      // 0..3
  const int lrow  = lane & 15;
  const int khalf = lane >> 4;             // 0 or 1
  const int mblk  = blockIdx.x << 4;

  int mload = mblk + lrow; if (mload >= M) mload = M - 1;   // clamp; store masked
  const __bf16* xrow = X + (size_t)mload * DIM;

  const int n0 = wv << 6;                  // 64-wide N slab per wave
  const __bf16* wr[4];
  v8f acc[4];
  #pragma unroll
  for (int t = 0; t < 4; ++t) {
    wr[t] = W + (size_t)(n0 + t * 16 + lrow) * DIM;
    const float bn = bias[n0 + t * 16 + lrow];
    #pragma unroll
    for (int i = 0; i < 8; ++i) acc[t][i] = bn;   // bias folded into C
  }

  #pragma unroll
  for (int k0 = 0; k0 < DIM; k0 += 32) {
    // A fragment: lane half splits K in groups of 8 within each 16 (ISA table)
    v8bf alo = *(const v8bf*)(xrow + k0 + (khalf << 3));
    v8bf ahi = *(const v8bf*)(xrow + k0 + 16 + (khalf << 3));
    v16bf a;
    #pragma unroll
    for (int i = 0; i < 8; ++i) { a[i] = alo[i]; a[i + 8] = ahi[i]; }

    // B fragments: same K split as A (lane half selects group of 8 within each 16)
    #pragma unroll
    for (int t = 0; t < 4; ++t) {
      v8bf blo = *(const v8bf*)(wr[t] + k0 + (khalf << 3));
      v8bf bhi = *(const v8bf*)(wr[t] + k0 + 16 + (khalf << 3));
      v16bf b;
      #pragma unroll
      for (int i = 0; i < 8; ++i) { b[i] = blo[i]; b[i + 8] = bhi[i]; }
      acc[t] = __builtin_amdgcn_wmma_f32_16x16x32_bf16(false, a, false, b,
                                                       (short)0, acc[t], false, false);
    }
  }

  const int mbase = mblk + (khalf << 3);    // C/D: lanes 16-31 hold M=8..15
  #pragma unroll
  for (int r = 0; r < 8; ++r) {
    int mr = mbase + r;
    if (mr < M) {
      #pragma unroll
      for (int t = 0; t < 4; ++t)
        Y[(size_t)mr * DIM + n0 + t * 16 + lrow] = acc[t][r];
    }
  }
}

// ---------------------------------------------------------------------------
// fact-level assembly / qualifier accumulation
// ---------------------------------------------------------------------------
__global__ __launch_bounds__(256)
void k_fact_assemble(const int* __restrict__ pri, const float* __restrict__ Phe,
                     const float* __restrict__ Pte, const float* __restrict__ Qprr,
                     float* __restrict__ fe2e, float* __restrict__ fr2r,
                     float* __restrict__ fe_cnt) {
  int f = blockIdx.x, d = threadIdx.x;
  int h = pri[f * 3 + 0], r = pri[f * 3 + 1], t = pri[f * 3 + 2];
  fe2e[(size_t)f * DIM + d] = Phe[(size_t)h * DIM + d] + Pte[(size_t)t * DIM + d];
  fr2r[(size_t)f * DIM + d] = Qprr[(size_t)r * DIM + d];
  if (d == 0) fe_cnt[f] = 2.0f;
}

__global__ __launch_bounds__(256)
void k_qual_accum(const int* __restrict__ qual, const int* __restrict__ q2f,
                  const float* __restrict__ Pqe, const float* __restrict__ Qqrr,
                  float* __restrict__ fe2e, float* __restrict__ fr2r,
                  float* __restrict__ fe_cnt) {
  int q = blockIdx.x, d = threadIdx.x;
  int fa = q2f[q];
  int qr = qual[q * 2 + 0], qe = qual[q * 2 + 1];
  atomicAdd(&fe2e[(size_t)fa * DIM + d], Pqe[(size_t)qe * DIM + d]);
  atomicAdd(&fr2r[(size_t)fa * DIM + d], Qqrr[(size_t)qr * DIM + d]);
  if (d == 0) atomicAdd(&fe_cnt[fa], 1.0f);
}

// X[row] = fe[row] - P[pri[row*3+col]]  (cast to bf16)
__global__ __launch_bounds__(256)
void k_build_x_fact(const float* __restrict__ fe, const float* __restrict__ P,
                    const int* __restrict__ pri, int col, __bf16* __restrict__ X) {
  int f = blockIdx.x, d = threadIdx.x;
  int i = pri[f * 3 + col];
  X[(size_t)f * DIM + d] = (__bf16)(fe[(size_t)f * DIM + d] - P[(size_t)i * DIM + d]);
}

// X[q] = fe[q2f[q]] - P[qual[q*2+col]]
__global__ __launch_bounds__(256)
void k_build_x_qual(const float* __restrict__ fe, const float* __restrict__ P,
                    const int* __restrict__ qual, int col, const int* __restrict__ q2f,
                    __bf16* __restrict__ X) {
  int q = blockIdx.x, d = threadIdx.x;
  int fa = q2f[q];
  int i = qual[q * 2 + col];
  X[(size_t)q * DIM + d] = (__bf16)(fe[(size_t)fa * DIM + d] - P[(size_t)i * DIM + d]);
}

// acc[idx] += S[row]; acc_cnt[idx] += fe_cnt[row] + off
__global__ __launch_bounds__(256)
void k_scatter_fact(const float* __restrict__ S, const int* __restrict__ pri, int col,
                    const float* __restrict__ fe_cnt, float off,
                    float* __restrict__ acc, float* __restrict__ acc_cnt) {
  int f = blockIdx.x, d = threadIdx.x;
  int i = pri[f * 3 + col];
  atomicAdd(&acc[(size_t)i * DIM + d], S[(size_t)f * DIM + d]);
  if (d == 0) atomicAdd(&acc_cnt[i], fe_cnt[f] + off);
}

__global__ __launch_bounds__(256)
void k_scatter_qual(const float* __restrict__ S, const int* __restrict__ qual, int col,
                    const int* __restrict__ q2f, const float* __restrict__ fe_cnt, float off,
                    float* __restrict__ acc, float* __restrict__ acc_cnt) {
  int q = blockIdx.x, d = threadIdx.x;
  int i = qual[q * 2 + col];
  atomicAdd(&acc[(size_t)i * DIM + d], S[(size_t)q * DIM + d]);
  if (d == 0) atomicAdd(&acc_cnt[i], fe_cnt[q2f[q]] + off);
}

// r2e: rel->ent mean contributions
__global__ __launch_bounds__(256)
void k_r2e_fact(const int* __restrict__ pri, const float* __restrict__ Qph,
                const float* __restrict__ Qpt, float* __restrict__ sum, float* __restrict__ cnt) {
  int f = blockIdx.x, d = threadIdx.x;
  int h = pri[f * 3 + 0], r = pri[f * 3 + 1], t = pri[f * 3 + 2];
  atomicAdd(&sum[(size_t)h * DIM + d], Qph[(size_t)r * DIM + d]);
  atomicAdd(&sum[(size_t)t * DIM + d], Qpt[(size_t)r * DIM + d]);
  if (d == 0) { atomicAdd(&cnt[h], 1.0f); atomicAdd(&cnt[t], 1.0f); }
}

__global__ __launch_bounds__(256)
void k_r2e_qual(const int* __restrict__ qual, const float* __restrict__ Qqq,
                float* __restrict__ sum, float* __restrict__ cnt) {
  int q = blockIdx.x, d = threadIdx.x;
  int qr = qual[q * 2 + 0], qe = qual[q * 2 + 1];
  atomicAdd(&sum[(size_t)qe * DIM + d], Qqq[(size_t)qr * DIM + d]);
  if (d == 0) atomicAdd(&cnt[qe], 1.0f);
}

// e2r: ent->rel mean contributions
__global__ __launch_bounds__(256)
void k_e2r_fact(const int* __restrict__ pri, const float* __restrict__ Php,
                const float* __restrict__ Ptp, float* __restrict__ sum, float* __restrict__ cnt) {
  int f = blockIdx.x, d = threadIdx.x;
  int h = pri[f * 3 + 0], r = pri[f * 3 + 1], t = pri[f * 3 + 2];
  atomicAdd(&sum[(size_t)r * DIM + d],
            Php[(size_t)h * DIM + d] + Ptp[(size_t)t * DIM + d]);
  if (d == 0) atomicAdd(&cnt[r], 2.0f);
}

__global__ __launch_bounds__(256)
void k_e2r_qual(const int* __restrict__ qual, const float* __restrict__ Pqq,
                float* __restrict__ sum, float* __restrict__ cnt) {
  int q = blockIdx.x, d = threadIdx.x;
  int qr = qual[q * 2 + 0], qe = qual[q * 2 + 1];
  atomicAdd(&sum[(size_t)qr * DIM + d], Pqq[(size_t)qe * DIM + d]);
  if (d == 0) atomicAdd(&cnt[qr], 1.0f);
}

// ---------------------------------------------------------------------------
// out[i] = LN(emb[i] + acc[i]/max(acc_cnt,1 per ref) + mean_sum[i]/max(mean_cnt,1))
// one block of 256 threads per row
// ---------------------------------------------------------------------------
__global__ __launch_bounds__(256)
void k_ln_out(const float* __restrict__ emb, const float* __restrict__ acc,
              const float* __restrict__ acc_cnt, const float* __restrict__ msum,
              const float* __restrict__ mcnt, const float* __restrict__ g,
              const float* __restrict__ b, float* __restrict__ out) {
  __shared__ float s1[DIM];
  __shared__ float s2[DIM];
  int i = blockIdx.x, d = threadIdx.x;
  float c1 = acc_cnt[i]; if (c1 == 0.0f) c1 = 1.0f;
  float c2 = mcnt[i];    float c2d = (c2 > 1.0f) ? c2 : 1.0f;  // seg_mean: sum==0 when cnt==0
  float x = emb[(size_t)i * DIM + d] + acc[(size_t)i * DIM + d] / c1 +
            msum[(size_t)i * DIM + d] / c2d;
  s1[d] = x; s2[d] = x * x;
  __syncthreads();
  for (int s = 128; s > 0; s >>= 1) {
    if (d < s) { s1[d] += s1[d + s]; s2[d] += s2[d + s]; }
    __syncthreads();
  }
  float m = s1[0] * (1.0f / DIM);
  float v = s2[0] * (1.0f / DIM) - m * m;
  out[(size_t)i * DIM + d] = (x - m) * rsqrtf(v + 1e-5f) * g[d] + b[d];
}

// ---------------------------------------------------------------------------
// host side
// ---------------------------------------------------------------------------
extern "C" void kernel_launch(void* const* d_in, const int* in_sizes, int n_in,
                              void* d_out, int out_size, void* d_ws, size_t ws_size,
                              hipStream_t stream) {
  (void)n_in; (void)out_size; (void)ws_size;
  const int NE = in_sizes[0] / DIM;       // 50000
  const int NR = in_sizes[1] / DIM;       // 500
  const int NF = in_sizes[38] / 3;        // 100000
  const int NQ = in_sizes[39];            // 50000

  const float* emb_ent = (const float*)d_in[0];
  const float* emb_rel = (const float*)d_in[1];
  // PROJ order: 0 he2e,1 te2e,2 qe2e,3 he2pr,4 te2pr,5 qe2qr,6 pr2he,7 pr2te,
  //             8 qr2qe,9 pr2r,10 qr2r,11 fe2he,12 fe2te,13 fe2qe,14 fr2pr,15 fr2qr
  const float* Wf[16]; const float* Bf[16];
  for (int i = 0; i < 16; ++i) { Wf[i] = (const float*)d_in[2 + 2 * i]; Bf[i] = (const float*)d_in[3 + 2 * i]; }
  const float* ln_ent_g = (const float*)d_in[34];
  const float* ln_ent_b = (const float*)d_in[35];
  const float* ln_rel_g = (const float*)d_in[36];
  const float* ln_rel_b = (const float*)d_in[37];
  const int* pri  = (const int*)d_in[38];
  const int* q2f  = (const int*)d_in[39];
  const int* qual = (const int*)d_in[40];
  float* out_ent = (float*)d_out;
  float* out_rel = (float*)d_out + (size_t)NE * DIM;

  // ---- carve workspace ----
  char* cur = (char*)d_ws;
  auto carve = [&](size_t bytes) -> void* {
    void* p = (void*)cur; cur += (bytes + 255) & ~(size_t)255; return p;
  };
  const size_t bE = (size_t)NE * DIM, bR = (size_t)NR * DIM, bF = (size_t)NF * DIM, bQ = (size_t)NQ * DIM;
  __bf16* Ebf = (__bf16*)carve(bE * 2);
  __bf16* Rbf = (__bf16*)carve(bR * 2);
  __bf16* Wbf[16];
  for (int i = 0; i < 16; ++i) Wbf[i] = (__bf16*)carve((size_t)DIM * DIM * 2);
  float* Pm[6];  for (int i = 0; i < 6; ++i) Pm[i] = (float*)carve(bE * 4);   // ent projections 0..5
  float* Qm[5];  for (int i = 0; i < 5; ++i) Qm[i] = (float*)carve(bR * 4);   // rel projections 6..10
  float* fe2e   = (float*)carve(bF * 4);
  float* fr2r   = (float*)carve(bF * 4);
  float* fe_cnt = (float*)carve((size_t)NF * 4);
  size_t bX = (bF > bQ ? bF : bQ);
  __bf16* Xbf = (__bf16*)carve(bX * 2);
  float*  S   = (float*)carve(bX * 4);
  float* e2e     = (float*)carve(bE * 4);
  float* e2e_cnt = (float*)carve((size_t)NE * 4);
  float* r2e_sum = (float*)carve(bE * 4);
  float* r2e_cnt = (float*)carve((size_t)NE * 4);
  float* r2r     = (float*)carve(bR * 4);
  float* r2r_cnt = (float*)carve((size_t)NR * 4);
  float* e2r_sum = (float*)carve(bR * 4);
  float* e2r_cnt = (float*)carve((size_t)NR * 4);

  // ---- zero accumulators (graph-capture-legal) ----
  hipMemsetAsync(e2e, 0, bE * 4, stream);
  hipMemsetAsync(e2e_cnt, 0, (size_t)NE * 4, stream);
  hipMemsetAsync(r2e_sum, 0, bE * 4, stream);
  hipMemsetAsync(r2e_cnt, 0, (size_t)NE * 4, stream);
  hipMemsetAsync(r2r, 0, bR * 4, stream);
  hipMemsetAsync(r2r_cnt, 0, (size_t)NR * 4, stream);
  hipMemsetAsync(e2r_sum, 0, bR * 4, stream);
  hipMemsetAsync(e2r_cnt, 0, (size_t)NR * 4, stream);

  // ---- bf16 conversions ----
  k_cvt_bf16<<<(int)((bE + 255) / 256), 256, 0, stream>>>(emb_ent, Ebf, (long long)bE);
  k_cvt_bf16<<<(int)((bR + 255) / 256), 256, 0, stream>>>(emb_rel, Rbf, (long long)bR);
  for (int i = 0; i < 16; ++i)
    k_cvt_bf16<<<(DIM * DIM + 255) / 256, 256, 0, stream>>>(Wf[i], Wbf[i], (long long)(DIM * DIM));

  // ---- first-level projections (WMMA GEMMs) ----
  auto gemm = [&](const __bf16* X, int w, float* Y, int M) {
    k_gemm_wmma_bf16<<<(M + 15) / 16, 128, 0, stream>>>(X, Wbf[w], Bf[w], Y, M);
  };
  for (int i = 0; i < 6; ++i) gemm(Ebf, i, Pm[i], NE);        // he2e..qe2qr
  for (int i = 0; i < 5; ++i) gemm(Rbf, 6 + i, Qm[i], NR);    // pr2he..qr2r

  // ---- fact-level message assembly ----
  k_fact_assemble<<<NF, 256, 0, stream>>>(pri, Pm[0], Pm[1], Qm[3], fe2e, fr2r, fe_cnt);
  k_qual_accum<<<NQ, 256, 0, stream>>>(qual, q2f, Pm[2], Qm[4], fe2e, fr2r, fe_cnt);

  // ---- second-level projections + scatters (reuse Xbf/S, serialized on stream) ----
  // S1: fe2he on (fe2e - he2e[heads]) -> e2e[heads], cnt += fe_cnt-1
  k_build_x_fact<<<NF, 256, 0, stream>>>(fe2e, Pm[0], pri, 0, Xbf);
  gemm(Xbf, 11, S, NF);
  k_scatter_fact<<<NF, 256, 0, stream>>>(S, pri, 0, fe_cnt, -1.0f, e2e, e2e_cnt);
  // S2: fe2te on (fe2e - te2e[tails]) -> e2e[tails]
  k_build_x_fact<<<NF, 256, 0, stream>>>(fe2e, Pm[1], pri, 2, Xbf);
  gemm(Xbf, 12, S, NF);
  k_scatter_fact<<<NF, 256, 0, stream>>>(S, pri, 2, fe_cnt, -1.0f, e2e, e2e_cnt);
  // S3: fe2qe on (fe2e[q2f] - qe2e[qual_ents]) -> e2e[qual_ents]
  k_build_x_qual<<<NQ, 256, 0, stream>>>(fe2e, Pm[2], qual, 1, q2f, Xbf);
  gemm(Xbf, 13, S, NQ);
  k_scatter_qual<<<NQ, 256, 0, stream>>>(S, qual, 1, q2f, fe_cnt, -1.0f, e2e, e2e_cnt);
  // S4: fr2pr on (fr2r - pr2r[rels]) -> r2r[rels], cnt += fr_cnt-1 = fe_cnt-2
  k_build_x_fact<<<NF, 256, 0, stream>>>(fr2r, Qm[3], pri, 1, Xbf);
  gemm(Xbf, 14, S, NF);
  k_scatter_fact<<<NF, 256, 0, stream>>>(S, pri, 1, fe_cnt, -2.0f, r2r, r2r_cnt);
  // S5: fr2qr on (fr2r[q2f] - qr2r[qual_rels]) -> r2r[qual_rels]
  k_build_x_qual<<<NQ, 256, 0, stream>>>(fr2r, Qm[4], qual, 0, q2f, Xbf);
  gemm(Xbf, 15, S, NQ);
  k_scatter_qual<<<NQ, 256, 0, stream>>>(S, qual, 0, q2f, fe_cnt, -2.0f, r2r, r2r_cnt);

  // ---- mean-aggregated cross messages ----
  k_r2e_fact<<<NF, 256, 0, stream>>>(pri, Qm[0], Qm[1], r2e_sum, r2e_cnt);
  k_r2e_qual<<<NQ, 256, 0, stream>>>(qual, Qm[2], r2e_sum, r2e_cnt);
  k_e2r_fact<<<NF, 256, 0, stream>>>(pri, Pm[3], Pm[4], e2r_sum, e2r_cnt);
  k_e2r_qual<<<NQ, 256, 0, stream>>>(qual, Pm[5], e2r_sum, e2r_cnt);

  // ---- residual + LayerNorm outputs ----
  k_ln_out<<<NE, 256, 0, stream>>>(emb_ent, e2e, e2e_cnt, r2e_sum, r2e_cnt,
                                   ln_ent_g, ln_ent_b, out_ent);
  k_ln_out<<<NR, 256, 0, stream>>>(emb_rel, r2r, r2r_cnt, e2r_sum, e2r_cnt,
                                   ln_rel_g, ln_rel_b, out_rel);
}
